// PhasedSNForecastModel_42606075576839
// MI455X (gfx1250) — compile-verified, matrix operand order checked by reading, not run on hardware
//
#include <hip/hip_runtime.h>
#include <hip/hip_bf16.h>
#include <stdint.h>

#define UNITS 1024
#define BATCH 256
#define TSTEPS 512
#define FEAT 16
#define OUT_STEPS 24
#define LEAK 0.001f

typedef __attribute__((ext_vector_type(16))) __bf16 v16bf;
typedef __attribute__((ext_vector_type(8)))  float v8f;

union FragU {
    uint4 q[2];
    v16bf v;
};

__device__ __forceinline__ v16bf load_frag(const unsigned short* lo, const unsigned short* hi) {
    FragU f;
    f.q[0] = *(const uint4*)lo;
    f.q[1] = *(const uint4*)hi;
    return f.v;
}

__device__ __forceinline__ unsigned short f2bf(float x) {
    unsigned int b = __float_as_uint(x);
    unsigned int r = b + 0x7fffu + ((b >> 16) & 1u);
    return (unsigned short)(r >> 16);
}

__device__ __forceinline__ float fast_tanh(float x) {
    x = fminf(fmaxf(x, -15.f), 15.f);
    float e = __expf(2.f * x);
    return (e - 1.f) / (e + 1.f);
}

__device__ __forceinline__ float fast_sigmoid(float x) {
    x = fminf(fmaxf(x, -30.f), 30.f);
    return 1.f / (1.f + __expf(-x));
}

// WMMA 16-bit fragment K-index for (lane, half j) — per CDNA5 ISA 16-bit A layout.
__device__ __forceinline__ int frag_k(int lane, int j) {
    int basek = (lane < 16) ? 0 : 8;
    return (j < 8) ? (basek + j) : (16 + basek + (j - 8));
}

// ---------------- packing kernels ----------------

// Bpack layout: [nt(256)][kt(32)][lane(32)][j(16)] bf16, nt = gate*64 + utile.
__global__ __launch_bounds__(256) void pack_rec_kernel(const float* __restrict__ rec,
                                                       unsigned short* __restrict__ Bpack) {
    int idx = blockIdx.x * 256 + threadIdx.x;
    if (idx >= 256 * 32 * 32 * 16) return;
    int j = idx & 15;
    int lane = (idx >> 4) & 31;
    int kt = (idx >> 9) & 31;
    int nt = idx >> 14;
    int k = kt * 32 + frag_k(lane, j);
    int col = nt * 16 + (lane & 15);
    Bpack[idx] = f2bf(rec[k * 4096 + col]);
}

// Kpack: [nt(256)][lane(32)][j(16)] bf16; K = 16 real + 16 zero pad.
__global__ __launch_bounds__(256) void pack_kern_kernel(const float* __restrict__ kern,
                                                        unsigned short* __restrict__ Kpack) {
    int idx = blockIdx.x * 256 + threadIdx.x;
    if (idx >= 256 * 32 * 16) return;
    int j = idx & 15;
    int lane = (idx >> 4) & 31;
    int nt = idx >> 9;
    int k = frag_k(lane, j);
    int col = nt * 16 + (lane & 15);
    Kpack[idx] = (k < FEAT) ? f2bf(kern[k * 4096 + col]) : (unsigned short)0;
}

// x (B,T,F) f32 -> xbf[t][b][f] bf16, tbuf[t][b] = x[b][t][0], maskbuf[t][b] = any(x != -1)
__global__ __launch_bounds__(256) void pack_x_kernel(const float* __restrict__ x,
                                                     unsigned short* __restrict__ xbf,
                                                     float* __restrict__ tbuf,
                                                     float* __restrict__ maskbuf) {
    int idx = blockIdx.x * 256 + threadIdx.x;
    if (idx >= TSTEPS * BATCH) return;
    int b = idx & (BATCH - 1);
    int t = idx >> 8;
    const float* row = x + ((size_t)b * TSTEPS + t) * FEAT;
    float any = 0.f;
    for (int f = 0; f < FEAT; ++f) {
        float v = row[f];
        if (v != -1.0f) any = 1.f;
        xbf[(size_t)(t * BATCH + b) * FEAT + f] = f2bf(v);
    }
    tbuf[t * BATCH + b] = row[0];
    maskbuf[t * BATCH + b] = any;
}

__global__ __launch_bounds__(256) void zero32_kernel(unsigned int* __restrict__ p, int n) {
    int i = blockIdx.x * 256 + threadIdx.x;
    if (i < n) p[i] = 0u;
}

// ---------------- fused PLSTM step ----------------
// Grid: 128 blocks x 256 threads = 1024 waves; wave tile = (mtile 0..15, utile 0..63).
__global__ __launch_bounds__(256) void plstm_step_kernel(
    const unsigned short* __restrict__ hbf_in,
    unsigned short* __restrict__ hbf_out,
    float* __restrict__ h, float* __restrict__ c,
    const unsigned short* __restrict__ Bpack,
    const unsigned short* __restrict__ Kpack,
    const unsigned short* __restrict__ xrowbf,   // [BATCH][FEAT] bf16
    const float* __restrict__ bias,
    const float* __restrict__ period,
    const float* __restrict__ phase,
    const float* __restrict__ ratio,
    const float* __restrict__ tptr, int tstride,
    const float* __restrict__ maskptr) {
    int tid = threadIdx.x;
    int lane = tid & 31;
    int wv = tid >> 5;
    int w = blockIdx.x * 8 + wv;
    int mtile = w & 15;
    int utile = w >> 4;   // 0..63

    v8f acc0 = {}, acc1 = {}, acc2 = {}, acc3 = {};

    const unsigned short* arow = hbf_in + (size_t)(mtile * 16 + (lane & 15)) * UNITS;
    int basek0 = (lane < 16) ? 0 : 8;

    // K-loop over rec_kernel: K = 1024 = 32 x 32
#pragma unroll 2
    for (int kt = 0; kt < 32; ++kt) {
        v16bf a = load_frag(arow + kt * 32 + basek0, arow + kt * 32 + basek0 + 16);
        const unsigned short* b0 = Bpack + ((size_t)((0 * 64 + utile) * 32 + kt) * 32 + lane) * 16;
        const unsigned short* b1 = Bpack + ((size_t)((1 * 64 + utile) * 32 + kt) * 32 + lane) * 16;
        const unsigned short* b2 = Bpack + ((size_t)((2 * 64 + utile) * 32 + kt) * 32 + lane) * 16;
        const unsigned short* b3 = Bpack + ((size_t)((3 * 64 + utile) * 32 + kt) * 32 + lane) * 16;
        acc0 = __builtin_amdgcn_wmma_f32_16x16x32_bf16(false, a, false, load_frag(b0, b0 + 8), (short)0, acc0, false, false);
        acc1 = __builtin_amdgcn_wmma_f32_16x16x32_bf16(false, a, false, load_frag(b1, b1 + 8), (short)0, acc1, false, false);
        acc2 = __builtin_amdgcn_wmma_f32_16x16x32_bf16(false, a, false, load_frag(b2, b2 + 8), (short)0, acc2, false, false);
        acc3 = __builtin_amdgcn_wmma_f32_16x16x32_bf16(false, a, false, load_frag(b3, b3 + 8), (short)0, acc3, false, false);
    }

    // x_t @ kernel contribution: one extra WMMA with K=16 zero-padded to 32
    {
        FragU fa;
        const unsigned short* xrow = xrowbf + (size_t)(mtile * 16 + (lane & 15)) * FEAT;
        fa.q[0] = *(const uint4*)(xrow + basek0);
        uint4 z; z.x = 0u; z.y = 0u; z.z = 0u; z.w = 0u;
        fa.q[1] = z;
        const unsigned short* k0 = Kpack + ((size_t)((0 * 64 + utile)) * 32 + lane) * 16;
        const unsigned short* k1 = Kpack + ((size_t)((1 * 64 + utile)) * 32 + lane) * 16;
        const unsigned short* k2 = Kpack + ((size_t)((2 * 64 + utile)) * 32 + lane) * 16;
        const unsigned short* k3 = Kpack + ((size_t)((3 * 64 + utile)) * 32 + lane) * 16;
        acc0 = __builtin_amdgcn_wmma_f32_16x16x32_bf16(false, fa.v, false, load_frag(k0, k0 + 8), (short)0, acc0, false, false);
        acc1 = __builtin_amdgcn_wmma_f32_16x16x32_bf16(false, fa.v, false, load_frag(k1, k1 + 8), (short)0, acc1, false, false);
        acc2 = __builtin_amdgcn_wmma_f32_16x16x32_bf16(false, fa.v, false, load_frag(k2, k2 + 8), (short)0, acc2, false, false);
        acc3 = __builtin_amdgcn_wmma_f32_16x16x32_bf16(false, fa.v, false, load_frag(k3, k3 + 8), (short)0, acc3, false, false);
    }

    // Gate + phased-time-gate math: fully lane-local.
    int n = lane & 15;
    int u = utile * 16 + n;
    float per = period[u], ph = phase[u], ro = ratio[u];
    float b_i = bias[u], b_f = bias[UNITS + u], b_g = bias[2 * UNITS + u], b_o = bias[3 * UNITS + u];
    int mbase = mtile * 16 + ((lane < 16) ? 0 : 8);

#pragma unroll
    for (int r = 0; r < 8; ++r) {
        int b = mbase + r;
        size_t off = (size_t)b * UNITS + u;
        float hold = h[off];
        float cold = c[off];
        float zi = acc0[r] + b_i;
        float zf = acc1[r] + b_f;
        float zg = acc2[r] + b_g;
        float zo = acc3[r] + b_o;
        float ig = fast_sigmoid(zi);
        float fg = fast_sigmoid(zf);
        float og = fast_sigmoid(zo);
        float nc = fg * cold + ig * fast_tanh(zg);
        float nh = og * fast_tanh(nc);

        float tv = tptr[b * tstride];
        float cr = fmodf(tv - ph, per);
        if (cr < 0.f) cr += per;
        cr /= per;
        float kup = 2.f * cr / ro;
        float kk = (cr < ro) ? (2.f - kup) : (LEAK * cr);
        kk = (cr < 0.5f * ro) ? kup : kk;

        float ho = kk * nh + (1.f - kk) * hold;
        float co = kk * nc + (1.f - kk) * cold;
        if (maskptr) {
            float m = maskptr[b];
            if (m == 0.f) { ho = hold; co = cold; }
        }
        h[off] = ho;
        c[off] = co;
        hbf_out[off] = f2bf(ho);
    }
}

// ---------------- dense head ----------------
__global__ __launch_bounds__(256) void dense_stack_kernel(
    const float* __restrict__ h,
    const float* __restrict__ w1, const float* __restrict__ b1,
    const float* __restrict__ w2, const float* __restrict__ b2,
    const float* __restrict__ w3, const float* __restrict__ b3,
    const float* __restrict__ wo, const float* __restrict__ bo,
    float* __restrict__ p, unsigned short* __restrict__ pbf,
    float* __restrict__ out, int step) {
    __shared__ float sh[UNITS];
    __shared__ float s1[256];
    __shared__ float s2[128];
    __shared__ float s3[64];
    int b = blockIdx.x;
    int t = threadIdx.x;
    for (int i = t; i < UNITS; i += 256) sh[i] = h[(size_t)b * UNITS + i];
    __syncthreads();
    {
        float acc = b1[t];
        for (int k = 0; k < UNITS; ++k) acc += sh[k] * w1[k * 256 + t];
        s1[t] = fast_tanh(acc);
    }
    __syncthreads();
    if (t < 128) {
        float acc = b2[t];
        for (int k = 0; k < 256; ++k) acc += s1[k] * w2[k * 128 + t];
        s2[t] = fast_tanh(acc);
    }
    __syncthreads();
    if (t < 64) {
        float acc = b3[t];
        for (int k = 0; k < 128; ++k) acc += s2[k] * w3[k * 64 + t];
        s3[t] = fast_tanh(acc);
    }
    __syncthreads();
    if (t < FEAT) {
        float acc = bo[t];
        for (int k = 0; k < 64; ++k) acc += s3[k] * wo[k * FEAT + t];
        p[b * FEAT + t] = acc;
        pbf[b * FEAT + t] = f2bf(acc);
        out[((size_t)b * OUT_STEPS + step) * FEAT + t] = acc;
    }
}

// ---------------- host orchestration ----------------
extern "C" void kernel_launch(void* const* d_in, const int* in_sizes, int n_in,
                              void* d_out, int out_size, void* d_ws, size_t ws_size,
                              hipStream_t stream) {
    (void)in_sizes; (void)n_in; (void)out_size; (void)ws_size;
    const float* x      = (const float*)d_in[0];
    const float* kern   = (const float*)d_in[1];
    const float* rec    = (const float*)d_in[2];
    const float* bias   = (const float*)d_in[3];
    const float* period = (const float*)d_in[4];
    const float* phase  = (const float*)d_in[5];
    const float* ratio  = (const float*)d_in[6];
    const float* w1     = (const float*)d_in[7];
    const float* b1     = (const float*)d_in[8];
    const float* w2     = (const float*)d_in[9];
    const float* b2     = (const float*)d_in[10];
    const float* w3     = (const float*)d_in[11];
    const float* b3     = (const float*)d_in[12];
    const float* wo     = (const float*)d_in[13];
    const float* bo     = (const float*)d_in[14];
    float* out = (float*)d_out;

    // workspace carve-up (256-byte aligned)
    char* ws = (char*)d_ws;
    size_t off = 0;
    auto carve = [&](size_t bytes) {
        char* p = ws + off;
        off += (bytes + 255) & ~(size_t)255;
        return p;
    };
    unsigned short* Bpack   = (unsigned short*)carve((size_t)256 * 32 * 32 * 16 * 2); // 8 MB
    unsigned short* Kpack   = (unsigned short*)carve((size_t)256 * 32 * 16 * 2);      // 256 KB
    unsigned short* xbf     = (unsigned short*)carve((size_t)TSTEPS * BATCH * FEAT * 2);
    float*          tbuf    = (float*)carve((size_t)TSTEPS * BATCH * 4);
    float*          maskbuf = (float*)carve((size_t)TSTEPS * BATCH * 4);
    float*          h       = (float*)carve((size_t)BATCH * UNITS * 4);
    float*          c       = (float*)carve((size_t)BATCH * UNITS * 4);
    unsigned short* hbf0    = (unsigned short*)carve((size_t)BATCH * UNITS * 2);
    unsigned short* hbf1    = (unsigned short*)carve((size_t)BATCH * UNITS * 2);
    float*          p       = (float*)carve((size_t)BATCH * FEAT * 4);
    unsigned short* pbf     = (unsigned short*)carve((size_t)BATCH * FEAT * 2);
    unsigned short* hbf[2] = { hbf0, hbf1 };

    // pack weights/inputs (deterministic, every call)
    pack_rec_kernel<<<(256 * 32 * 32 * 16 + 255) / 256, 256, 0, stream>>>(rec, Bpack);
    pack_kern_kernel<<<(256 * 32 * 16 + 255) / 256, 256, 0, stream>>>(kern, Kpack);
    pack_x_kernel<<<(TSTEPS * BATCH + 255) / 256, 256, 0, stream>>>(x, xbf, tbuf, maskbuf);

    // zero state
    zero32_kernel<<<(BATCH * UNITS + 255) / 256, 256, 0, stream>>>((unsigned int*)h, BATCH * UNITS);
    zero32_kernel<<<(BATCH * UNITS + 255) / 256, 256, 0, stream>>>((unsigned int*)c, BATCH * UNITS);
    zero32_kernel<<<(BATCH * UNITS / 2 + 255) / 256, 256, 0, stream>>>((unsigned int*)hbf0, BATCH * UNITS / 2);
    zero32_kernel<<<(BATCH * UNITS / 2 + 255) / 256, 256, 0, stream>>>((unsigned int*)hbf1, BATCH * UNITS / 2);

    int cur = 0;
    // warm-up recurrence: 512 masked PLSTM steps
    for (int t = 0; t < TSTEPS; ++t) {
        plstm_step_kernel<<<128, 256, 0, stream>>>(
            hbf[cur], hbf[cur ^ 1], h, c, Bpack, Kpack,
            xbf + (size_t)t * BATCH * FEAT, bias, period, phase, ratio,
            tbuf + t * BATCH, 1, maskbuf + t * BATCH);
        cur ^= 1;
    }
    // pred0
    dense_stack_kernel<<<BATCH, 256, 0, stream>>>(h, w1, b1, w2, b2, w3, b3, wo, bo, p, pbf, out, 0);
    // autoregressive steps
    for (int s = 1; s < OUT_STEPS; ++s) {
        plstm_step_kernel<<<128, 256, 0, stream>>>(
            hbf[cur], hbf[cur ^ 1], h, c, Bpack, Kpack,
            pbf, bias, period, phase, ratio,
            p, FEAT, (const float*)nullptr);
        cur ^= 1;
        dense_stack_kernel<<<BATCH, 256, 0, stream>>>(h, w1, b1, w2, b2, w3, b3, wo, bo, p, pbf, out, s);
    }
}
